// WeatherSTGNN_65000035058473
// MI455X (gfx1250) — matrix operand order repaired and checked by reading, not compiled, not run on hardware
//
#include <hip/hip_runtime.h>
#include <hip/hip_bf16.h>

// ---------------------------------------------------------------------------
// WeatherSTGNN for MI455X (gfx1250, wave32, WMMA)
// B=4096, T=24, N=7, IN=11 (6 dyn + 5 stat), H=64, FUT=48
// Each block: 8 batch elements -> 56 rows (padded to 64 = 4 x 16-row tiles),
// 4 waves, entire 72-step recurrence in one kernel, activations in LDS.
// Gate phase processes one 16-column group at a time (i/f/g/o tiles together)
// to keep live accumulators at 32 VGPRs and avoid VGPR-MSB switching.
// ---------------------------------------------------------------------------

typedef __attribute__((ext_vector_type(16))) _Float16 v16h;
typedef __attribute__((ext_vector_type(8)))  float    v8f;

#define BATCH   4096
#define TT      24
#define NNODE   7
#define NIN     11
#define NDYN    6
#define HH      64
#define FUTS    48
#define BPB     8          // batch elements per block
#define ROWS    64         // padded rows (56 real)
#define RROWS   (BPB*NNODE)

// packed B-fragment images in d_ws (dwords). Each fragment = 32 lanes x 8 dwords.
#define FR_WIH  0          // 16 ntiles x 2 khalves  (w_ih^T : 64 x 256)
#define FR_WHH  32         // 16 x 2                 (w_hh^T : 64 x 256)
#define FR_G1   64         // 4 x 2                  (gcn1_w : 64 x 64)
#define FR_G2   72         // 4 x 2                  (gcn2_w : 64 x 64)
#define FR_ENC  80         // 4 x 1 (K padded 11->32)(enc_w  : 11 x 64)
#define FR_D1   84         // 2 x 2                  (dec_w1 : 64 x 32)
#define N_FRAGS 88
#define FRAG_DW 256
#define TOT_DW  (N_FRAGS * FRAG_DW)

__device__ inline float sigm_f(float x)  { return 1.f / (1.f + __expf(-x)); }
__device__ inline float tanh_f(float x)  { return 1.f - 2.f / (__expf(2.f * x) + 1.f); }

__device__ inline v8f wmma16(v16h a, v16h b, v8f c) {
  return __builtin_amdgcn_wmma_f32_16x16x32_f16(false, a, false, b, (short)0, c, false, false);
}

// A-matrix 16x32 f16 fragment from f32 LDS tile (documented CDNA5 layout):
// lanes 0-15: M=lane, v0..3 = K pairs 0..7, v4..7 = K pairs 16..23
// lanes 16-31: same M, +8 on the K pair base.
__device__ inline v16h load_a_f32(const float* X, int ld, int m0, int k0, int kmax, int lane) {
  int m  = m0 + (lane & 15);
  int hi = lane >> 4;
  v16h a;
#pragma unroll
  for (int v = 0; v < 8; ++v) {
    int kk = k0 + ((v < 4) ? (2 * v + 8 * hi) : (16 + 2 * (v - 4) + 8 * hi));
    float x0 = (kk     < kmax) ? X[m * ld + kk]     : 0.f;
    float x1 = (kk + 1 < kmax) ? X[m * ld + kk + 1] : 0.f;
    a[2 * v]     = (_Float16)x0;
    a[2 * v + 1] = (_Float16)x1;
  }
  return a;
}

// B fragment: pre-packed by prep kernel; lane reads 32 contiguous bytes.
__device__ inline v16h load_b_frag(const unsigned* frags, int f, int lane) {
  union { uint4 q[2]; v16h h; } u;
  const uint4* p = (const uint4*)(frags + f * FRAG_DW + lane * 8);
  u.q[0] = p[0];
  u.q[1] = p[1];
  return u.h;
}

// GEMM: (16 x 64) @ (64 x 64) tile into dst (raw, no bias)
__device__ inline void gemm_h64(const float* src, float* dst, const unsigned* frags,
                                int fbase, int m0, int lane) {
  int nl = lane & 15, hi = lane >> 4;
  v16h a0 = load_a_f32(src, HH, m0, 0, HH, lane);
  v16h a1 = load_a_f32(src, HH, m0, 32, HH, lane);
#pragma unroll
  for (int nt = 0; nt < 4; ++nt) {
    v8f acc = {};
    acc = wmma16(a0, load_b_frag(frags, fbase + nt * 2, lane), acc);
    acc = wmma16(a1, load_b_frag(frags, fbase + nt * 2 + 1, lane), acc);
#pragma unroll
    for (int r = 0; r < 8; ++r)
      dst[(m0 + r + 8 * hi) * HH + nt * 16 + nl] = acc[r];
  }
}

// ---------------------------------------------------------------------------
// Prep: pack all weights into WMMA B-fragment images (f16) in d_ws.
// ---------------------------------------------------------------------------
__global__ void prep_frags(const float* __restrict__ w_ih, const float* __restrict__ w_hh,
                           const float* __restrict__ g1,  const float* __restrict__ g2,
                           const float* __restrict__ encw, const float* __restrict__ d1w,
                           unsigned* __restrict__ ws) {
  int dw = blockIdx.x * blockDim.x + threadIdx.x;
  if (dw >= TOT_DW) return;
  int frag = dw >> 8;
  int rem  = dw & 255;
  int lane = rem >> 3;
  int v    = rem & 7;
  int nl   = lane & 15;
  int k2   = ((lane >> 4) * 8) + 2 * v + ((v >= 4) ? 8 : 0);     // K pair base (table order: 8h+2v, 16+8h+2(v-4))
  float x0 = 0.f, x1 = 0.f;
  if (frag < FR_WHH) {                       // w_ih^T: B[k][n] = w_ih[n][k]
    int f = frag - FR_WIH;
    int n = (f >> 1) * 16 + nl, k = (f & 1) * 32 + k2;
    x0 = w_ih[n * HH + k]; x1 = w_ih[n * HH + k + 1];
  } else if (frag < FR_G1) {                 // w_hh^T
    int f = frag - FR_WHH;
    int n = (f >> 1) * 16 + nl, k = (f & 1) * 32 + k2;
    x0 = w_hh[n * HH + k]; x1 = w_hh[n * HH + k + 1];
  } else if (frag < FR_G2) {                 // gcn1_w: B[k][n] = g1[k][n]
    int f = frag - FR_G1;
    int n = (f >> 1) * 16 + nl, k = (f & 1) * 32 + k2;
    x0 = g1[k * HH + n]; x1 = g1[(k + 1) * HH + n];
  } else if (frag < FR_ENC) {                // gcn2_w
    int f = frag - FR_G2;
    int n = (f >> 1) * 16 + nl, k = (f & 1) * 32 + k2;
    x0 = g2[k * HH + n]; x1 = g2[(k + 1) * HH + n];
  } else if (frag < FR_D1) {                 // enc_w, K padded 11 -> 32
    int f = frag - FR_ENC;
    int n = f * 16 + nl, k = k2;
    x0 = (k     < NIN) ? encw[k * HH + n]       : 0.f;
    x1 = (k + 1 < NIN) ? encw[(k + 1) * HH + n] : 0.f;
  } else {                                   // dec_w1: 64 x 32
    int f = frag - FR_D1;
    int n = (f >> 1) * 16 + nl, k = (f & 1) * 32 + k2;
    x0 = d1w[k * 32 + n]; x1 = d1w[(k + 1) * 32 + n];
  }
  union { _Float16 h[2]; unsigned u; } p;
  p.h[0] = (_Float16)x0; p.h[1] = (_Float16)x1;
  ws[dw] = p.u;
}

// ---------------------------------------------------------------------------
// Main persistent kernel: full 72-step recurrence per block.
// ---------------------------------------------------------------------------
__global__ __launch_bounds__(128) void stgnn_main(
    const float* __restrict__ xh, const float* __restrict__ adj,
    const float* __restrict__ encb, const float* __restrict__ g1b,
    const float* __restrict__ g2b,  const float* __restrict__ bih,
    const float* __restrict__ bhh,  const float* __restrict__ d1b,
    const float* __restrict__ dw2,  const float* __restrict__ db2,
    const unsigned* __restrict__ frags, float* __restrict__ out) {

  __shared__ float s_hx[ROWS * HH];
  __shared__ float s_cx[ROWS * HH];
  __shared__ float s_b1[ROWS * HH];
  __shared__ float s_b2[ROWS * HH];
  __shared__ float s_xt[ROWS * 16];
  __shared__ float s_adjn[BPB * NNODE * NNODE];
  __shared__ float s_pred[ROWS * NDYN];
  __shared__ float s_stat[ROWS * 5];
  __shared__ float s_encb[HH], s_g1b[HH], s_g2b[HH], s_bsum[4 * HH];
  __shared__ float s_d1b[32], s_db2[8], s_dw2[32 * NDYN];
  __shared__ float s_dv[64];

  const int tid = threadIdx.x;
  const int lane = tid & 31;
  const int wid = tid >> 5;
  const int m0 = wid * 16;
  const int b0 = blockIdx.x * BPB;
  const int nl = lane & 15, hi = lane >> 4;

  // ---- init state / constants ----
  for (int i = tid; i < ROWS * HH; i += 128) { s_hx[i] = 0.f; s_cx[i] = 0.f; }
  for (int i = tid; i < HH; i += 128) { s_encb[i] = encb[i]; s_g1b[i] = g1b[i]; s_g2b[i] = g2b[i]; }
  for (int i = tid; i < 4 * HH; i += 128) s_bsum[i] = bih[i] + bhh[i];
  if (tid < 32) s_d1b[tid] = d1b[tid];
  if (tid < NDYN) s_db2[tid] = db2[tid];
  for (int i = tid; i < 32 * NDYN; i += 128) s_dw2[i] = dw2[i];

  for (int i = tid; i < ROWS * NDYN; i += 128) {
    int r = i / NDYN, c = i % NDYN;
    float v = 0.f;
    if (r < RROWS) {
      int lb = r / NNODE, n = r % NNODE;
      v = xh[(size_t)(b0 + lb) * (TT * NNODE * NIN) + (TT - 1) * (NNODE * NIN) + n * NIN + c];
    }
    s_pred[i] = v;
  }
  for (int i = tid; i < ROWS * 5; i += 128) {
    int r = i / 5, c = i % 5;
    float v = 0.f;
    if (r < RROWS) {
      int lb = r / NNODE, n = r % NNODE;
      v = xh[(size_t)(b0 + lb) * (TT * NNODE * NIN) + (TT - 1) * (NNODE * NIN) + n * NIN + NDYN + c];
    }
    s_stat[i] = v;
  }
  // normalized adjacency (diag forced to 1, D^-1/2 A D^-1/2)
  for (int r = tid; r < RROWS; r += 128) {
    int lb = r / NNODE, i0 = r % NNODE;
    float s = 0.f;
#pragma unroll
    for (int j = 0; j < NNODE; ++j) {
      float a = (j == i0) ? 1.f : adj[(size_t)(b0 + lb) * (NNODE * NNODE) + i0 * NNODE + j];
      s += a;
    }
    s_dv[r] = rsqrtf(fmaxf(s, 1.f));
  }
  __syncthreads();
  for (int idx = tid; idx < BPB * NNODE * NNODE; idx += 128) {
    int lb = idx / (NNODE * NNODE);
    int ij = idx % (NNODE * NNODE);
    int i0 = ij / NNODE, j = ij % NNODE;
    float a = (i0 == j) ? 1.f : adj[(size_t)(b0 + lb) * (NNODE * NNODE) + i0 * NNODE + j];
    s_adjn[idx] = s_dv[lb * NNODE + i0] * a * s_dv[lb * NNODE + j];
  }
  __syncthreads();

  // ---- recurrence over 24 history + 48 future steps ----
  for (int t = 0; t < TT + FUTS; ++t) {
    // 1. build x_t (ROWS x 16, cols 11..15 zero)
    for (int i = tid; i < ROWS * 16; i += 128) {
      int r = i >> 4, c = i & 15;
      float v = 0.f;
      if (r < RROWS && c < NIN) {
        if (t < TT) {
          int lb = r / NNODE, n = r % NNODE;
          v = xh[(size_t)(b0 + lb) * (TT * NNODE * NIN) + t * (NNODE * NIN) + n * NIN + c];
        } else {
          v = (c < NDYN) ? s_pred[r * NDYN + c] : s_stat[r * 5 + (c - NDYN)];
        }
      }
      s_xt[i] = v;
    }
    __syncthreads();

    // 2. encoder: relu(x_t @ enc_w + enc_b)  (K=11 zero-padded to 32)
    {
      v16h ax = load_a_f32(s_xt, 16, m0, 0, 16, lane);
#pragma unroll
      for (int nt = 0; nt < 4; ++nt) {
        v8f acc = {};
        acc = wmma16(ax, load_b_frag(frags, FR_ENC + nt, lane), acc);
#pragma unroll
        for (int r = 0; r < 8; ++r) {
          int col = nt * 16 + nl;
          s_b1[(m0 + r + 8 * hi) * HH + col] = fmaxf(acc[r] + s_encb[col], 0.f);
        }
      }
    }
    // 3. gcn1 pre-mix GEMM (own-tile rows, no sync needed after enc)
    gemm_h64(s_b1, s_b2, frags, FR_G1, m0, lane);
    __syncthreads();
    // 4. adjacency mix + bias + relu -> s_b1 (pad rows zeroed)
    for (int idx = tid; idx < RROWS * HH; idx += 128) {
      int row = idx >> 6, h = idx & 63;
      int lb = row / NNODE, i0 = row - lb * NNODE;
      const float* ar = s_adjn + lb * (NNODE * NNODE) + i0 * NNODE;
      float s = 0.f;
#pragma unroll
      for (int j = 0; j < NNODE; ++j) s += ar[j] * s_b2[(lb * NNODE + j) * HH + h];
      s_b1[idx] = fmaxf(s + s_g1b[h], 0.f);
    }
    for (int idx = tid; idx < (ROWS - RROWS) * HH; idx += 128) s_b1[RROWS * HH + idx] = 0.f;
    __syncthreads();
    // 5. gcn2
    gemm_h64(s_b1, s_b2, frags, FR_G2, m0, lane);
    __syncthreads();
    for (int idx = tid; idx < RROWS * HH; idx += 128) {
      int row = idx >> 6, h = idx & 63;
      int lb = row / NNODE, i0 = row - lb * NNODE;
      const float* ar = s_adjn + lb * (NNODE * NNODE) + i0 * NNODE;
      float s = 0.f;
#pragma unroll
      for (int j = 0; j < NNODE; ++j) s += ar[j] * s_b2[(lb * NNODE + j) * HH + h];
      s_b1[idx] = fmaxf(s + s_g2b[h], 0.f);
    }
    for (int idx = tid; idx < (ROWS - RROWS) * HH; idx += 128) s_b1[RROWS * HH + idx] = 0.f;
    __syncthreads();

    // 6. LSTM gates: g = s @ w_ih^T + hx @ w_hh^T
    //    One 16-column group per iteration; i/f/g/o accumulated together
    //    (tiles tq, 4+tq, 8+tq, 12+tq) -> only 32 live accumulator VGPRs.
    {
      v16h as0 = load_a_f32(s_b1, HH, m0, 0, HH, lane);
      v16h as1 = load_a_f32(s_b1, HH, m0, 32, HH, lane);
      v16h ah0 = load_a_f32(s_hx, HH, m0, 0, HH, lane);
      v16h ah1 = load_a_f32(s_hx, HH, m0, 32, HH, lane);
#pragma unroll
      for (int tq = 0; tq < 4; ++tq) {
        v8f gi = {}, gf = {}, gg = {}, go = {};
        gi = wmma16(as0, load_b_frag(frags, FR_WIH + (tq)      * 2,     lane), gi);
        gi = wmma16(as1, load_b_frag(frags, FR_WIH + (tq)      * 2 + 1, lane), gi);
        gi = wmma16(ah0, load_b_frag(frags, FR_WHH + (tq)      * 2,     lane), gi);
        gi = wmma16(ah1, load_b_frag(frags, FR_WHH + (tq)      * 2 + 1, lane), gi);
        gf = wmma16(as0, load_b_frag(frags, FR_WIH + (4 + tq)  * 2,     lane), gf);
        gf = wmma16(as1, load_b_frag(frags, FR_WIH + (4 + tq)  * 2 + 1, lane), gf);
        gf = wmma16(ah0, load_b_frag(frags, FR_WHH + (4 + tq)  * 2,     lane), gf);
        gf = wmma16(ah1, load_b_frag(frags, FR_WHH + (4 + tq)  * 2 + 1, lane), gf);
        gg = wmma16(as0, load_b_frag(frags, FR_WIH + (8 + tq)  * 2,     lane), gg);
        gg = wmma16(as1, load_b_frag(frags, FR_WIH + (8 + tq)  * 2 + 1, lane), gg);
        gg = wmma16(ah0, load_b_frag(frags, FR_WHH + (8 + tq)  * 2,     lane), gg);
        gg = wmma16(ah1, load_b_frag(frags, FR_WHH + (8 + tq)  * 2 + 1, lane), gg);
        go = wmma16(as0, load_b_frag(frags, FR_WIH + (12 + tq) * 2,     lane), go);
        go = wmma16(as1, load_b_frag(frags, FR_WIH + (12 + tq) * 2 + 1, lane), go);
        go = wmma16(ah0, load_b_frag(frags, FR_WHH + (12 + tq) * 2,     lane), go);
        go = wmma16(ah1, load_b_frag(frags, FR_WHH + (12 + tq) * 2 + 1, lane), go);
        int col = tq * 16 + nl;
#pragma unroll
        for (int r = 0; r < 8; ++r) {
          int row = m0 + r + 8 * hi;
          float iv = sigm_f(gi[r] + s_bsum[col]);
          float fv = sigm_f(gf[r] + s_bsum[HH + col]);
          float gv = tanh_f(gg[r] + s_bsum[2 * HH + col]);
          float ov = sigm_f(go[r] + s_bsum[3 * HH + col]);
          float cv = fv * s_cx[row * HH + col] + iv * gv;
          float hv = ov * tanh_f(cv);
          s_cx[row * HH + col] = cv;
          s_hx[row * HH + col] = hv;
        }
      }
    }

    // 7. decoder on future steps
    if (t >= TT) {
      // d1 = relu(hx @ dec_w1 + dec_b1) -> s_b2[:, 0:32] (own-tile rows)
      v16h da0 = load_a_f32(s_hx, HH, m0, 0, HH, lane);
      v16h da1 = load_a_f32(s_hx, HH, m0, 32, HH, lane);
#pragma unroll
      for (int nt = 0; nt < 2; ++nt) {
        v8f acc = {};
        acc = wmma16(da0, load_b_frag(frags, FR_D1 + nt * 2,     lane), acc);
        acc = wmma16(da1, load_b_frag(frags, FR_D1 + nt * 2 + 1, lane), acc);
#pragma unroll
        for (int r = 0; r < 8; ++r) {
          int col = nt * 16 + nl;
          s_b2[(m0 + r + 8 * hi) * HH + col] = fmaxf(acc[r] + s_d1b[col], 0.f);
        }
      }
      // res = tanh(d1 @ dec_w2 + dec_b2) * 0.05 ; pred = clip(pred + res, 0, 1)
      for (int idx = lane; idx < 16 * NDYN; idx += 32) {
        int r = idx / NDYN, c = idx % NDYN;
        int row = m0 + r;
        if (row < RROWS) {
          float s = s_db2[c];
#pragma unroll
          for (int k = 0; k < 32; ++k) s += s_b2[row * HH + k] * s_dw2[k * NDYN + c];
          float p = s_pred[row * NDYN + c] + tanh_f(s) * 0.05f;
          p = fminf(fmaxf(p, 0.f), 1.f);
          s_pred[row * NDYN + c] = p;
          int lb = row / NNODE, nn = row % NNODE;
          out[(((size_t)(b0 + lb) * FUTS + (t - TT)) * NNODE + nn) * NDYN + c] = p;
        }
      }
    }
    __syncthreads();
  }
}

extern "C" void kernel_launch(void* const* d_in, const int* in_sizes, int n_in,
                              void* d_out, int out_size, void* d_ws, size_t ws_size,
                              hipStream_t stream) {
  const float* xh   = (const float*)d_in[0];
  const float* adj  = (const float*)d_in[1];
  const float* encw = (const float*)d_in[2];
  const float* encb = (const float*)d_in[3];
  const float* g1w  = (const float*)d_in[4];
  const float* g1b  = (const float*)d_in[5];
  const float* g2w  = (const float*)d_in[6];
  const float* g2b  = (const float*)d_in[7];
  const float* wih  = (const float*)d_in[8];
  const float* whh  = (const float*)d_in[9];
  const float* bih  = (const float*)d_in[10];
  const float* bhh  = (const float*)d_in[11];
  const float* d1w  = (const float*)d_in[12];
  const float* d1b  = (const float*)d_in[13];
  const float* d2w  = (const float*)d_in[14];
  const float* d2b  = (const float*)d_in[15];
  unsigned* wsu = (unsigned*)d_ws;
  float* out = (float*)d_out;
  (void)in_sizes; (void)n_in; (void)out_size; (void)ws_size;

  prep_frags<<<(TOT_DW + 255) / 256, 256, 0, stream>>>(wih, whh, g1w, g2w, encw, d1w, wsu);
  stgnn_main<<<BATCH / BPB, 128, 0, stream>>>(xh, adj, encb, g1b, g2b, bih, bhh,
                                              d1b, d2w, d2b, wsu, out);
}
